// GATEdgeClassifier_89721866813769
// MI455X (gfx1250) — matrix operand order repaired and checked by reading, not compiled, not run on hardware
//
#include <hip/hip_runtime.h>

// ---------------- problem constants ----------------
#define NN     50000      // nodes
#define NE     800000     // edges
#define NET    (NE + NN)  // edges + self loops
#define D0     128        // emb dim
#define NHID   32
#define NHEADS 8

typedef __attribute__((ext_vector_type(16))) _Float16 v16h;
typedef __attribute__((ext_vector_type(8)))  _Float16 v8h;
typedef __attribute__((ext_vector_type(8)))  float    v8f;

// monotone float<->uint key for atomicMax on floats
__device__ __forceinline__ unsigned fkey(float f) {
    unsigned b = __float_as_uint(f);
    return (b & 0x80000000u) ? ~b : (b | 0x80000000u);
}
__device__ __forceinline__ float funkey(unsigned k) {
    unsigned b = (k & 0x80000000u) ? (k & 0x7fffffffu) : ~k;
    return __uint_as_float(b);
}
#define KEY_NEG_INF 0x007FFFFFu   // fkey(-inf)

__device__ __forceinline__ v8f wmma_f16(v16h a, v16h b, v8f c) {
    return __builtin_amdgcn_wmma_f32_16x16x32_f16(
        false, a, false, b, (short)0, c, false, false);
}

// build A fragment (16x32 f16, row-major source row pointer) per ISA layout
__device__ __forceinline__ v16h ld_a_frag(const _Float16* __restrict__ row, int k0, int lane) {
    int ka = k0 + ((lane >> 4) << 3);           // 0 or 8
    v8h a0 = *(const v8h*)(row + ka);           // K = ka..ka+7
    v8h a1 = *(const v8h*)(row + ka + 16);      // K = ka+16..ka+23
    v16h a;
#pragma unroll
    for (int i = 0; i < 8; i++) { a[i] = a0[i]; a[i + 8] = a1[i]; }
    return a;
}
// build B fragment from pre-transposed B (Nc x K row-major) per ISA layout
__device__ __forceinline__ v16h ld_b_frag(const _Float16* __restrict__ row, int k0, int lane) {
    int kb = k0 + ((lane >> 4) << 3);           // 0 or 8
    v8h b0 = *(const v8h*)(row + kb);
    v8h b1 = *(const v8h*)(row + kb + 16);
    v16h b;
#pragma unroll
    for (int i = 0; i < 8; i++) { b[i] = b0[i]; b[i + 8] = b1[i]; }
    return b;
}

// ---------------- utility kernels ----------------
__global__ void fill_f32(float* p, float v, long long n) {
    long long t = blockIdx.x * (long long)blockDim.x + threadIdx.x;
    if (t < n) p[t] = v;
}
__global__ void fill_u32(unsigned* p, unsigned v, long long n) {
    long long t = blockIdx.x * (long long)blockDim.x + threadIdx.x;
    if (t < n) p[t] = v;
}
__global__ void k_f2h(const float* __restrict__ in, _Float16* __restrict__ out, long long n) {
    long long t = blockIdx.x * (long long)blockDim.x + threadIdx.x;
    if (t < n) out[t] = (_Float16)in[t];
}
// W (K x hc, row-major f32) -> Wt (hc x K, row-major f16)
__global__ void k_transpose16(const float* __restrict__ W, _Float16* __restrict__ Wt, int K, int hc) {
    long long t = blockIdx.x * (long long)blockDim.x + threadIdx.x;
    if (t >= (long long)K * hc) return;
    int n = (int)(t / K), k = (int)(t % K);
    Wt[t] = (_Float16)W[(size_t)k * hc + n];
}
__global__ void k_embed(const int* __restrict__ x, const float* __restrict__ emb, float* __restrict__ h) {
    long long t = blockIdx.x * (long long)blockDim.x + threadIdx.x;
    if (t >= (long long)NN * D0) return;
    int n = (int)(t >> 7), d = (int)(t & 127);
    h[t] = emb[(size_t)x[n] * D0 + d];
}
// edge_attr column means (sum phase)
__global__ void k_ea_sum(const float* __restrict__ ea, float* __restrict__ sums) {
    __shared__ float sh[256 * 3];
    float s0 = 0.f, s1 = 0.f, s2 = 0.f;
    for (long long e = blockIdx.x * (long long)blockDim.x + threadIdx.x; e < NE;
         e += (long long)gridDim.x * blockDim.x) {
        s0 += ea[e * 3 + 0]; s1 += ea[e * 3 + 1]; s2 += ea[e * 3 + 2];
    }
    sh[threadIdx.x] = s0; sh[256 + threadIdx.x] = s1; sh[512 + threadIdx.x] = s2;
    __syncthreads();
    for (int off = 128; off > 0; off >>= 1) {
        if ((int)threadIdx.x < off) {
            sh[threadIdx.x]       += sh[threadIdx.x + off];
            sh[256 + threadIdx.x] += sh[256 + threadIdx.x + off];
            sh[512 + threadIdx.x] += sh[512 + threadIdx.x + off];
        }
        __syncthreads();
    }
    if (threadIdx.x == 0) {
        atomicAdd(&sums[0], sh[0]); atomicAdd(&sums[1], sh[256]); atomicAdd(&sums[2], sh[512]);
    }
}
__global__ void k_ea_fin(float* s) {
    if (threadIdx.x < 3) s[threadIdx.x] *= (1.0f / (float)NE);
}
// veA[h][j] = sum_k We[j, h*out+k]*ae[h,k];  aeself[h] = ea_mean . veA[h]
__global__ void k_prep(const float* __restrict__ We, const float* __restrict__ ae,
                       const float* __restrict__ eam, float* __restrict__ veA,
                       float* __restrict__ aeself, int heads, int out, int hc) {
    int h = threadIdx.x;
    if (h >= heads) return;
    float vs[3];
    for (int j = 0; j < 3; j++) {
        float s = 0.f;
        for (int k = 0; k < out; k++) s += We[(size_t)j * hc + h * out + k] * ae[h * out + k];
        veA[h * 3 + j] = s; vs[j] = s;
    }
    aeself[h] = eam[0] * vs[0] + eam[1] * vs[1] + eam[2] * vs[2];
}

// ---------------- WMMA GEMM: C[M x Nc] = A[M x K] * Bt^T, f16 in / f32 out ----
// grid = (M/16, Nc/16), block = 32 (one wave per 16x16 tile)
__global__ void k_gemm(const _Float16* __restrict__ A, const _Float16* __restrict__ Bt,
                       float* __restrict__ C, int K, int Nc) {
    int lane = threadIdx.x;
    int r16 = lane & 15, hi = lane >> 4;
    const _Float16* arow = A + ((size_t)blockIdx.x * 16 + r16) * K;
    const _Float16* brow = Bt + ((size_t)blockIdx.y * 16 + r16) * K;
    v8f c = {};
    for (int k0 = 0; k0 < K; k0 += 32) {
        v16h a = ld_a_frag(arow, k0, lane);
        v16h b = ld_b_frag(brow, k0, lane);
        c = wmma_f16(a, b, c);
    }
    float* crow = C + ((size_t)blockIdx.x * 16 + hi * 8) * Nc + (size_t)blockIdx.y * 16 + r16;
#pragma unroll
    for (int r = 0; r < 8; r++) crow[(size_t)r * Nc] = c[r];
}

// ---------------- attention kernels ----------------
__global__ void k_alpha(const float* __restrict__ xs, const float* __restrict__ a_s,
                        const float* __restrict__ a_d, float* __restrict__ as_o,
                        float* __restrict__ ad_o, int heads, int out, int hc) {
    long long t = blockIdx.x * (long long)blockDim.x + threadIdx.x;
    if (t >= (long long)NN * heads) return;
    int n = (int)(t / heads), h = (int)(t % heads);
    const float* row = xs + (size_t)n * hc + h * out;
    const float* vs = a_s + h * out;
    const float* vd = a_d + h * out;
    float ss = 0.f, sd = 0.f;
    for (int k = 0; k < out; k++) { float v = row[k]; ss += v * vs[k]; sd += v * vd[k]; }
    as_o[t] = ss; ad_o[t] = sd;
}

__global__ void k_logits(const int* __restrict__ src, const int* __restrict__ dst,
                         const float* __restrict__ ea, const float* __restrict__ veA,
                         const float* __restrict__ aeself, const float* __restrict__ as_,
                         const float* __restrict__ ad_, float* __restrict__ elog,
                         unsigned* __restrict__ mkey, int heads) {
    long long t = blockIdx.x * (long long)blockDim.x + threadIdx.x;
    if (t >= (long long)NET * heads) return;
    int e = (int)(t / heads), h = (int)(t % heads);
    int s, d; float aE;
    if (e < NE) {
        s = src[e]; d = dst[e];
        const float* p = ea + (size_t)e * 3;
        aE = p[0] * veA[h * 3] + p[1] * veA[h * 3 + 1] + p[2] * veA[h * 3 + 2];
    } else { s = d = e - NE; aE = aeself[h]; }
    float l = as_[(size_t)s * heads + h] + ad_[(size_t)d * heads + h] + aE;
    l = l > 0.f ? l : 0.2f * l;                    // leaky_relu(0.2)
    elog[t] = l;
    atomicMax(mkey + (size_t)d * heads + h, fkey(l));
}

__global__ void k_exp(const int* __restrict__ dst, float* __restrict__ elog,
                      const unsigned* __restrict__ mkey, float* __restrict__ den, int heads) {
    long long t = blockIdx.x * (long long)blockDim.x + threadIdx.x;
    if (t >= (long long)NET * heads) return;
    int e = (int)(t / heads), h = (int)(t % heads);
    int d = (e < NE) ? dst[e] : e - NE;
    float m = funkey(mkey[(size_t)d * heads + h]);
    float ex = expf(elog[t] - m);
    elog[t] = ex;
    atomicAdd(den + (size_t)d * heads + h, ex);
}

__global__ void k_agg(const int* __restrict__ src, const int* __restrict__ dst,
                      const float* __restrict__ elog, const float* __restrict__ den,
                      const float* __restrict__ xs, float* __restrict__ agg,
                      int heads, int out, int hc) {
    long long t = blockIdx.x * (long long)blockDim.x + threadIdx.x;
    if (t >= (long long)NET * heads) return;
    int e = (int)(t / heads), h = (int)(t % heads);
    int s = (e < NE) ? src[e] : e - NE;
    int d = (e < NE) ? dst[e] : e - NE;
    float alpha = elog[t] / (den[(size_t)d * heads + h] + 1e-16f);
    const float* xr = xs + (size_t)s * hc + h * out;
    float* ar = agg + (size_t)d * hc + h * out;
    for (int k = 0; k < out; k++) atomicAdd(ar + k, xr[k] * alpha);
}

__global__ void k_bias_elu(float* __restrict__ h, const float* __restrict__ b,
                           int hc, long long n) {
    long long t = blockIdx.x * (long long)blockDim.x + threadIdx.x;
    if (t >= n) return;
    float v = h[t] + b[(int)(t % hc)];
    h[t] = v > 0.f ? v : expm1f(v);
}

// ---------------- fused edge MLP: WMMA GEMM + ReLU + 32x3 projection ----------
// block = 64 (2 waves), grid = NE/16.  wave w computes C cols [16w,16w+16).
__global__ void k_mlp(const _Float16* __restrict__ h2, const int* __restrict__ src,
                      const int* __restrict__ dst, const _Float16* __restrict__ mw1t, // 32 x 64
                      const float* __restrict__ mb1, const float* __restrict__ mw2,   // 32 x 3
                      const float* __restrict__ mb2, float* __restrict__ out) {
    __shared__ float tile[16 * 32];
    int lane = threadIdx.x & 31, w = threadIdx.x >> 5;
    int r16 = lane & 15, hi = lane >> 4;
    int e = blockIdx.x * 16 + r16;
    int ng = w * 16 + r16;
    const _Float16* brow = mw1t + (size_t)ng * 64;
    v8f c = {};
    { // K 0..31 : h2[src]
        const _Float16* arow = h2 + (size_t)src[e] * 32;
        c = wmma_f16(ld_a_frag(arow, 0, lane), ld_b_frag(brow, 0, lane), c);
    }
    { // K 32..63 : h2[dst]
        const _Float16* arow = h2 + (size_t)dst[e] * 32;
        c = wmma_f16(ld_a_frag(arow, 0, lane), ld_b_frag(brow, 32, lane), c);
    }
    float bias = mb1[ng];
#pragma unroll
    for (int r = 0; r < 8; r++) {
        float v = c[r] + bias;
        tile[(hi * 8 + r) * 32 + ng] = v > 0.f ? v : 0.f;   // relu
    }
    __syncthreads();
    if (threadIdx.x < 48) {
        int el = threadIdx.x / 3, j = threadIdx.x % 3;
        float s = mb2[j];
        const float* row = tile + el * 32;
        for (int k = 0; k < 32; k++) s += row[k] * mw2[k * 3 + j];
        out[((size_t)blockIdx.x * 16 + el) * 3 + j] = s;
    }
}

// ---------------- host-side orchestration ----------------
static inline unsigned nb(long long n, int t) { return (unsigned)((n + t - 1) / t); }

static void run_layer(hipStream_t st, const float* hin, float* hout, int K, int heads, int out,
                      const float* W, const float* a_s, const float* a_d, const float* We,
                      const float* a_e, const float* bias,
                      _Float16* h16, _Float16* wt16, float* xs, float* as_, float* ad_,
                      unsigned* mkey, float* den, float* elog,
                      const float* ea_mean, float* veA, float* aeself,
                      const int* src, const int* dst, const float* eattr) {
    int hc = heads * out;
    const int T = 256;
    k_transpose16<<<nb((long long)hc * K, T), T, 0, st>>>(W, wt16, K, hc);
    k_f2h<<<nb((long long)NN * K, T), T, 0, st>>>(hin, h16, (long long)NN * K);
    dim3 g(NN / 16, hc / 16);
    k_gemm<<<g, 32, 0, st>>>(h16, wt16, xs, K, hc);
    k_prep<<<1, 64, 0, st>>>(We, a_e, ea_mean, veA, aeself, heads, out, hc);
    k_alpha<<<nb((long long)NN * heads, T), T, 0, st>>>(xs, a_s, a_d, as_, ad_, heads, out, hc);
    fill_u32<<<nb((long long)NN * heads, T), T, 0, st>>>(mkey, KEY_NEG_INF, (long long)NN * heads);
    fill_f32<<<nb((long long)NN * heads, T), T, 0, st>>>(den, 0.f, (long long)NN * heads);
    fill_f32<<<nb((long long)NN * hc, T), T, 0, st>>>(hout, 0.f, (long long)NN * hc);
    long long nt = (long long)NET * heads;
    k_logits<<<nb(nt, T), T, 0, st>>>(src, dst, eattr, veA, aeself, as_, ad_, elog, mkey, heads);
    k_exp<<<nb(nt, T), T, 0, st>>>(dst, elog, mkey, den, heads);
    k_agg<<<nb(nt, T), T, 0, st>>>(src, dst, elog, den, xs, hout, heads, out, hc);
    k_bias_elu<<<nb((long long)NN * hc, T), T, 0, st>>>(hout, bias, hc, (long long)NN * hc);
}

extern "C" void kernel_launch(void* const* d_in, const int* in_sizes, int n_in,
                              void* d_out, int out_size, void* d_ws, size_t ws_size,
                              hipStream_t stream) {
    const int*   x     = (const int*)d_in[0];
    const int*   ei    = (const int*)d_in[1];
    const float* eattr = (const float*)d_in[2];
    const float* emb   = (const float*)d_in[3];
    const float *w0 = (const float*)d_in[4],  *as0 = (const float*)d_in[5],
                *ad0 = (const float*)d_in[6], *we0 = (const float*)d_in[7],
                *ae0 = (const float*)d_in[8], *b0  = (const float*)d_in[9];
    const float *w1 = (const float*)d_in[10], *as1 = (const float*)d_in[11],
                *ad1 = (const float*)d_in[12],*we1 = (const float*)d_in[13],
                *ae1 = (const float*)d_in[14],*b1  = (const float*)d_in[15];
    const float *w2 = (const float*)d_in[16], *as2 = (const float*)d_in[17],
                *ad2 = (const float*)d_in[18],*we2 = (const float*)d_in[19],
                *ae2 = (const float*)d_in[20],*b2  = (const float*)d_in[21];
    const float *mw1 = (const float*)d_in[22],*mb1 = (const float*)d_in[23],
                *mw2 = (const float*)d_in[24],*mb2 = (const float*)d_in[25];

    const int* src = ei;        // edge_index[0]
    const int* dst = ei + NE;   // edge_index[1]

    // workspace layout (256B aligned)
    char* ws = (char*)d_ws;
    size_t o = 0;
    auto alloc = [&](size_t bytes) { size_t r = o; o = (o + bytes + 255) & ~(size_t)255; return r; };
    float*    bufA  = (float*)(ws + alloc((size_t)NN * 256 * 4));
    float*    bufB  = (float*)(ws + alloc((size_t)NN * 256 * 4));
    _Float16* h16   = (_Float16*)(ws + alloc((size_t)NN * 256 * 2));
    float*    xs    = (float*)(ws + alloc((size_t)NN * 256 * 4));
    float*    as_   = (float*)(ws + alloc((size_t)NN * 8 * 4));
    float*    ad_   = (float*)(ws + alloc((size_t)NN * 8 * 4));
    unsigned* mkey  = (unsigned*)(ws + alloc((size_t)NN * 8 * 4));
    float*    den   = (float*)(ws + alloc((size_t)NN * 8 * 4));
    float*    elog  = (float*)(ws + alloc((size_t)NET * 8 * 4));
    _Float16* wt16  = (_Float16*)(ws + alloc((size_t)256 * 256 * 2));
    _Float16* mw1t  = (_Float16*)(ws + alloc((size_t)32 * 64 * 2));
    float*    small = (float*)(ws + alloc(256));
    float* ea_mean = small;        // 4 floats
    float* veA     = small + 4;    // 24 floats
    float* aeself  = small + 28;   // 8 floats

    const int T = 256;
    // node embedding
    k_embed<<<nb((long long)NN * D0, T), T, 0, stream>>>(x, emb, bufA);
    // edge_attr mean (for self-loop edges)
    fill_f32<<<1, 32, 0, stream>>>(ea_mean, 0.f, 4);
    k_ea_sum<<<256, 256, 0, stream>>>(eattr, ea_mean);
    k_ea_fin<<<1, 32, 0, stream>>>(ea_mean);

    // three GAT layers (ping-pong bufA/bufB)
    run_layer(stream, bufA, bufB, 128, NHEADS, NHID, w0, as0, ad0, we0, ae0, b0,
              h16, wt16, xs, as_, ad_, mkey, den, elog, ea_mean, veA, aeself, src, dst, eattr);
    run_layer(stream, bufB, bufA, 256, NHEADS, NHID, w1, as1, ad1, we1, ae1, b1,
              h16, wt16, xs, as_, ad_, mkey, den, elog, ea_mean, veA, aeself, src, dst, eattr);
    run_layer(stream, bufA, bufB, 256, 1, NHID, w2, as2, ad2, we2, ae2, b2,
              h16, wt16, xs, as_, ad_, mkey, den, elog, ea_mean, veA, aeself, src, dst, eattr);

    // final node features (N x 32) -> f16 for the fused edge MLP (reuse h16 buffer)
    _Float16* h2_16 = h16;
    k_f2h<<<nb((long long)NN * 32, T), T, 0, stream>>>(bufB, h2_16, (long long)NN * 32);
    k_transpose16<<<nb((long long)32 * 64, T), T, 0, stream>>>(mw1, mw1t, 64, 32);
    k_mlp<<<NE / 16, 64, 0, stream>>>(h2_16, src, dst, mw1t, mb1, mw2, mb2, (float*)d_out);
}
